// SparseAttention_74148315398287
// MI455X (gfx1250) — matrix operand order repaired and checked by reading, not compiled, not run on hardware
//
#include <hip/hip_runtime.h>
#include <hip/hip_bf16.h>
#include <math.h>

typedef _Float16 half_t;
typedef __attribute__((ext_vector_type(16))) _Float16 v16h;
typedef __attribute__((ext_vector_type(8)))  _Float16 v8h;
typedef __attribute__((ext_vector_type(4)))  _Float16 v4h;
typedef __attribute__((ext_vector_type(8)))  float    v8f;
typedef __attribute__((ext_vector_type(4)))  int      v4i;

#define DIMD   512
#define NHEAD  8
#define HDIM   64
#define SBLK   64
#define SFACT  4
#define BATCH  4
#define SEQ    4096
#define NBLK   (SEQ / SBLK)        /* 64 */
#define NSTR   (NBLK / SFACT)      /* 16 */
#define MTOT   (BATCH * SEQ)       /* 16384 */
#define QKV_ELEMS ((size_t)BATCH * NHEAD * SEQ * HDIM)  /* 8388608 */
#define QSCALE 0.125f              /* HD^-0.5 */

#if __has_builtin(__builtin_amdgcn_global_load_async_to_lds_b128)
#define ASYNC_COPY 1
#else
#define ASYNC_COPY 0
#endif

typedef __attribute__((address_space(1))) v4i glob_v4i;   // prints as "int4 __device__*"
typedef __attribute__((address_space(3))) v4i lds_v4i;    // prints as "int4 __shared__*"

__device__ __forceinline__ int lane_id() { return (int)(threadIdx.x & 31); }

__device__ __forceinline__ void async_wait_all() {
#if ASYNC_COPY
#if __has_builtin(__builtin_amdgcn_s_wait_asynccnt)
  __builtin_amdgcn_s_wait_asynccnt(0);
#else
  asm volatile("s_wait_asynccnt 0x0" ::: "memory");
#endif
#endif
}

// copy 4096 contiguous halves (8KB) global -> LDS
__device__ __forceinline__ void tile_copy_4k(const half_t* __restrict__ src,
                                             half_t* dst, int tid) {
#if ASYNC_COPY
#pragma unroll
  for (int it = 0; it < 4; ++it) {
    const int i = tid * 8 + it * 1024;
    __builtin_amdgcn_global_load_async_to_lds_b128(
        (glob_v4i*)(src + i), (lds_v4i*)(dst + i), 0, 0);
  }
#else
  v8h tmp[4];
#pragma unroll
  for (int it = 0; it < 4; ++it) tmp[it] = *(const v8h*)(src + tid * 8 + it * 1024);
#pragma unroll
  for (int it = 0; it < 4; ++it) *(v8h*)(dst + tid * 8 + it * 1024) = tmp[it];
#endif
}

__device__ __forceinline__ v8f wmma_f16(v16h a, v16h b, v8f c) {
  return __builtin_amdgcn_wmma_f32_16x16x32_f16(false, a, false, b, (short)0, c,
                                                false, false);
}

// A fragment (16x32, f16) from LDS tile, row stride lda halves.
__device__ __forceinline__ v16h lds_load_a(const half_t* A, int lda) {
  const int lane = lane_id();
  const int row  = lane & 15;
  const int kb   = (lane >> 4) << 4;           // 0 or 16
  const half_t* p = A + row * lda + kb;
  v8h lo = *(const v8h*)(p);
  v8h hi = *(const v8h*)(p + 8);
  v16h a;
#pragma unroll
  for (int i = 0; i < 8; ++i) { a[i] = lo[i]; a[i + 8] = hi[i]; }
  return a;
}

// B fragment (32(K)x16(N), f16) from K-contiguous LDS tile Bt[col][k].
__device__ __forceinline__ v16h lds_load_bt(const half_t* Bt, int ldb) {
  const int lane = lane_id();
  const int col  = lane & 15;
  const int kb   = (lane >> 4) << 4;           // 0 or 16
  return *(const v16h*)(Bt + col * ldb + kb);
}

__device__ __forceinline__ float halfgrp_max(float x) {
#pragma unroll
  for (int m = 1; m < 16; m <<= 1) x = fmaxf(x, __shfl_xor(x, m, 32));
  return x;
}
__device__ __forceinline__ float halfgrp_sum(float x) {
#pragma unroll
  for (int m = 1; m < 16; m <<= 1) x += __shfl_xor(x, m, 32);
  return x;
}

// ---------------------------------------------------------------------------
// Kernel 1: qkv = x @ Wqkv, store q (scaled), k, v as f16 in [B,H,L,HD]
// ---------------------------------------------------------------------------
__global__ __launch_bounds__(128) void sattn_qkv_kernel(
    const float* __restrict__ x, const float* __restrict__ Wqkv,
    half_t* __restrict__ qw, half_t* __restrict__ kw, half_t* __restrict__ vw) {
  __shared__ __align__(32) half_t As[64 * 32];
  __shared__ __align__(32) half_t Bs[64 * 32];
  const int tid = threadIdx.x;
  const int w   = tid >> 5;
  const int n0  = blockIdx.x * 64;   // column tile in [0,1536)
  const int m0  = blockIdx.y * 64;   // row tile in [0,16384)

  v8f acc[4];
#pragma unroll
  for (int c = 0; c < 4; ++c)
#pragma unroll
    for (int j = 0; j < 8; ++j) acc[c][j] = 0.0f;

  for (int kk = 0; kk < DIMD; kk += 32) {
    // ---- batched global loads (A rows + B columns) ----
    float4 fa[4], fb[4];
#pragma unroll
    for (int it = 0; it < 4; ++it) {
      const int i = tid * 4 + it * 512;
      const int m = i >> 5, k = i & 31;
      fa[it] = *(const float4*)(x + (size_t)(m0 + m) * DIMD + kk + k);
    }
#pragma unroll
    for (int it = 0; it < 4; ++it) {
      const int i = tid * 4 + it * 512;
      const int n = i & 63, k = i >> 6;
      fb[it] = *(const float4*)(Wqkv + (size_t)(kk + k) * (3 * DIMD) + n0 + n);
    }
    // ---- convert + LDS stores ----
#pragma unroll
    for (int it = 0; it < 4; ++it) {
      const int i = tid * 4 + it * 512;
      const int m = i >> 5, k = i & 31;
      v4h hv = {(half_t)fa[it].x, (half_t)fa[it].y, (half_t)fa[it].z, (half_t)fa[it].w};
      *(v4h*)(As + m * 32 + k) = hv;
    }
#pragma unroll
    for (int it = 0; it < 4; ++it) {
      const int i = tid * 4 + it * 512;
      const int n = i & 63, k = i >> 6;
      Bs[(n + 0) * 32 + k] = (half_t)fb[it].x;
      Bs[(n + 1) * 32 + k] = (half_t)fb[it].y;
      Bs[(n + 2) * 32 + k] = (half_t)fb[it].z;
      Bs[(n + 3) * 32 + k] = (half_t)fb[it].w;
    }
    __syncthreads();
    v16h a = lds_load_a(As + w * 16 * 32, 32);
#pragma unroll
    for (int c = 0; c < 4; ++c) {
      v16h b = lds_load_bt(Bs + c * 16 * 32, 32);
      acc[c] = wmma_f16(a, b, acc[c]);
    }
    __syncthreads();
  }

  // epilogue: n0 is 64-aligned -> whole tile maps to one of q/k/v and one head
  const int sec = n0 >> 9;                // 0=q 1=k 2=v
  const int h   = (n0 & 511) >> 6;
  half_t* dst   = (sec == 0) ? qw : (sec == 1) ? kw : vw;
  const float mul = (sec == 0) ? QSCALE : 1.0f;
  const int ccol  = lane_id() & 15;
  const int rbase = (lane_id() >> 4) << 3;
#pragma unroll
  for (int c = 0; c < 4; ++c)
#pragma unroll
    for (int r = 0; r < 8; ++r) {
      const int m  = m0 + w * 16 + rbase + r;
      const int d  = c * 16 + ccol;
      const int bb = m >> 12, l = m & 4095;
      dst[(((size_t)(bb * NHEAD + h) * SEQ + l) << 6) + d] =
          (half_t)(acc[c][r] * mul);
    }
}

// stage one key block: Ks row-major [t][d] (async), Vt transposed [d][t]
__device__ __forceinline__ void sattn_stage_kv(const half_t* __restrict__ ksrc,
                                               const half_t* __restrict__ vsrc,
                                               half_t* Ks, half_t* Vt, int tid) {
  tile_copy_4k(ksrc, Ks, tid);
  v8h vv[4];
#pragma unroll
  for (int it = 0; it < 4; ++it)
    vv[it] = *(const v8h*)(vsrc + tid * 8 + it * 1024);
#pragma unroll
  for (int it = 0; it < 4; ++it) {
    const int i = tid * 8 + it * 1024;
    const int t = i >> 6, d0 = i & 63;
#pragma unroll
    for (int j = 0; j < 8; ++j) Vt[(d0 + j) * 64 + t] = vv[it][j];
  }
}

// ---------------------------------------------------------------------------
// Kernel 2: per (b,h,block) local + strided-global attention, blended 0.5/0.5
// ---------------------------------------------------------------------------
__global__ __launch_bounds__(128) void sattn_attn_kernel(
    const half_t* __restrict__ qw, const half_t* __restrict__ kw,
    const half_t* __restrict__ vw, half_t* __restrict__ cw) {
  __shared__ __align__(32) half_t Qs[64 * 64];
  __shared__ __align__(32) half_t Ks[64 * 64];
  __shared__ __align__(32) half_t Vt[64 * 64];
  __shared__ __align__(32) half_t Ps[64 * 64];
  const int tid = threadIdx.x;
  const int w   = tid >> 5;
  const int blk = blockIdx.x;
  const int nb  = blk % NBLK;
  const int h   = (blk / NBLK) % NHEAD;
  const int bb  = blk / (NBLK * NHEAD);
  const size_t headoff = (size_t)(bb * NHEAD + h) * SEQ * HDIM;
  const int ccol  = lane_id() & 15;
  const int rbase = (lane_id() >> 4) << 3;

  // stage Q (scaled q) and own-block K/V
  tile_copy_4k(qw + headoff + (size_t)nb * SBLK * HDIM, Qs, tid);
  sattn_stage_kv(kw + headoff + (size_t)nb * SBLK * HDIM,
                 vw + headoff + (size_t)nb * SBLK * HDIM, Ks, Vt, tid);
  async_wait_all();
  __syncthreads();

  // ---------------- phase L: exact local attention over own block ----------
  v8f s[4];
#pragma unroll
  for (int c = 0; c < 4; ++c)
#pragma unroll
    for (int j = 0; j < 8; ++j) s[c][j] = 0.0f;
#pragma unroll
  for (int kkk = 0; kkk < 2; ++kkk) {
    v16h a = lds_load_a(Qs + w * 16 * 64 + kkk * 32, 64);
#pragma unroll
    for (int c = 0; c < 4; ++c)
      s[c] = wmma_f16(a, lds_load_bt(Ks + c * 16 * 64 + kkk * 32, 64), s[c]);
  }
#pragma unroll
  for (int r = 0; r < 8; ++r) {
    float mx = fmaxf(fmaxf(s[0][r], s[1][r]), fmaxf(s[2][r], s[3][r]));
    mx = halfgrp_max(mx);
    float sum = 0.0f;
#pragma unroll
    for (int c = 0; c < 4; ++c) {
      const float p = __expf(s[c][r] - mx);
      s[c][r] = p;
      sum += p;
    }
    sum = halfgrp_sum(sum);
    const float inv = 1.0f / sum;
#pragma unroll
    for (int c = 0; c < 4; ++c) s[c][r] *= inv;
  }
#pragma unroll
  for (int c = 0; c < 4; ++c)
#pragma unroll
    for (int r = 0; r < 8; ++r)
      Ps[(w * 16 + rbase + r) * 64 + c * 16 + ccol] = (half_t)s[c][r];
  // per-wave LDS ops are in-order: A-frag reload of own rows is safe
  v8f ol[4];
#pragma unroll
  for (int c = 0; c < 4; ++c)
#pragma unroll
    for (int j = 0; j < 8; ++j) ol[c][j] = 0.0f;
#pragma unroll
  for (int kkk = 0; kkk < 2; ++kkk) {
    v16h a = lds_load_a(Ps + w * 16 * 64 + kkk * 32, 64);
#pragma unroll
    for (int c = 0; c < 4; ++c)
      ol[c] = wmma_f16(a, lds_load_bt(Vt + c * 16 * 64 + kkk * 32, 64), ol[c]);
  }

  // ---------------- phase G: flash over strided blocks (+self if not dup) ---
  v8f acc[4];
#pragma unroll
  for (int c = 0; c < 4; ++c)
#pragma unroll
    for (int j = 0; j < 8; ++j) acc[c][j] = 0.0f;
  float mrow[8], lrow[8];
#pragma unroll
  for (int r = 0; r < 8; ++r) { mrow[r] = -INFINITY; lrow[r] = 0.0f; }

  const int nkb = (nb % SFACT == 0) ? NSTR : NSTR + 1;
  for (int it = 0; it < nkb; ++it) {
    const int jb = (it < NSTR) ? it * SFACT : nb;
    __syncthreads();                         // prior AV reads of Ks/Vt done
    sattn_stage_kv(kw + headoff + (size_t)jb * SBLK * HDIM,
                   vw + headoff + (size_t)jb * SBLK * HDIM, Ks, Vt, tid);
    if (it + 1 < NSTR) {                     // hint next strided block
      __builtin_prefetch(kw + headoff + (size_t)(it + 1) * SFACT * SBLK * HDIM, 0, 0);
      __builtin_prefetch(vw + headoff + (size_t)(it + 1) * SFACT * SBLK * HDIM, 0, 0);
    }
    async_wait_all();
    __syncthreads();

    v8f sg[4];
#pragma unroll
    for (int c = 0; c < 4; ++c)
#pragma unroll
      for (int j = 0; j < 8; ++j) sg[c][j] = 0.0f;
#pragma unroll
    for (int kkk = 0; kkk < 2; ++kkk) {
      v16h a = lds_load_a(Qs + w * 16 * 64 + kkk * 32, 64);
#pragma unroll
      for (int c = 0; c < 4; ++c)
        sg[c] = wmma_f16(a, lds_load_bt(Ks + c * 16 * 64 + kkk * 32, 64), sg[c]);
    }
#pragma unroll
    for (int r = 0; r < 8; ++r) {
      float bm = fmaxf(fmaxf(sg[0][r], sg[1][r]), fmaxf(sg[2][r], sg[3][r]));
      bm = halfgrp_max(bm);
      const float mnew  = fmaxf(mrow[r], bm);
      const float alpha = __expf(mrow[r] - mnew);
      mrow[r] = mnew;
      float sum = 0.0f;
#pragma unroll
      for (int c = 0; c < 4; ++c) {
        const float p = __expf(sg[c][r] - mnew);
        sg[c][r] = p;
        sum += p;
      }
      sum = halfgrp_sum(sum);
      lrow[r] = lrow[r] * alpha + sum;
#pragma unroll
      for (int c = 0; c < 4; ++c) acc[c][r] *= alpha;
    }
#pragma unroll
    for (int c = 0; c < 4; ++c)
#pragma unroll
      for (int r = 0; r < 8; ++r)
        Ps[(w * 16 + rbase + r) * 64 + c * 16 + ccol] = (half_t)sg[c][r];
#pragma unroll
    for (int kkk = 0; kkk < 2; ++kkk) {
      v16h a = lds_load_a(Ps + w * 16 * 64 + kkk * 32, 64);
#pragma unroll
      for (int c = 0; c < 4; ++c)
        acc[c] = wmma_f16(a, lds_load_bt(Vt + c * 16 * 64 + kkk * 32, 64), acc[c]);
    }
  }

  // blend + store context [B, L, H*HD] as f16
#pragma unroll
  for (int r = 0; r < 8; ++r) {
    const float invl = 1.0f / lrow[r];
    const int row = nb * 64 + w * 16 + rbase + r;
#pragma unroll
    for (int c = 0; c < 4; ++c) {
      const float o = 0.5f * (ol[c][r] + acc[c][r] * invl);
      cw[(size_t)(bb * SEQ + row) * DIMD + h * HDIM + c * 16 + ccol] = (half_t)o;
    }
  }
}

// ---------------------------------------------------------------------------
// Kernel 3: out = ctx @ Wproj + bproj   (fp32 output)
// ---------------------------------------------------------------------------
__global__ __launch_bounds__(128) void sattn_proj_kernel(
    const half_t* __restrict__ cw, const float* __restrict__ Wproj,
    const float* __restrict__ bproj, float* __restrict__ out) {
  __shared__ __align__(32) half_t As[64 * 32];
  __shared__ __align__(32) half_t Bs[64 * 32];
  const int tid = threadIdx.x;
  const int w   = tid >> 5;
  const int n0  = blockIdx.x * 64;
  const int m0  = blockIdx.y * 64;

  v8f acc[4];
#pragma unroll
  for (int c = 0; c < 4; ++c)
#pragma unroll
    for (int j = 0; j < 8; ++j) acc[c][j] = 0.0f;

  for (int kk = 0; kk < DIMD; kk += 32) {
    // A tile: f16 rows, 8-half contiguous chunks -> async copy to LDS
#if ASYNC_COPY
#pragma unroll
    for (int it = 0; it < 2; ++it) {
      const int i = tid * 8 + it * 1024;
      const int m = i >> 5, k = i & 31;
      __builtin_amdgcn_global_load_async_to_lds_b128(
          (glob_v4i*)(cw + (size_t)(m0 + m) * DIMD + kk + k),
          (lds_v4i*)(As + i), 0, 0);
    }
#else
    {
      v8h ta[2];
#pragma unroll
      for (int it = 0; it < 2; ++it) {
        const int i = tid * 8 + it * 1024;
        const int m = i >> 5, k = i & 31;
        ta[it] = *(const v8h*)(cw + (size_t)(m0 + m) * DIMD + kk + k);
      }
#pragma unroll
      for (int it = 0; it < 2; ++it) *(v8h*)(As + tid * 8 + it * 1024) = ta[it];
    }
#endif
    // B tile transposed, fp32 -> f16
    float4 fb[4];
#pragma unroll
    for (int it = 0; it < 4; ++it) {
      const int i = tid * 4 + it * 512;
      const int n = i & 63, k = i >> 6;
      fb[it] = *(const float4*)(Wproj + (size_t)(kk + k) * DIMD + n0 + n);
    }
#pragma unroll
    for (int it = 0; it < 4; ++it) {
      const int i = tid * 4 + it * 512;
      const int n = i & 63, k = i >> 6;
      Bs[(n + 0) * 32 + k] = (half_t)fb[it].x;
      Bs[(n + 1) * 32 + k] = (half_t)fb[it].y;
      Bs[(n + 2) * 32 + k] = (half_t)fb[it].z;
      Bs[(n + 3) * 32 + k] = (half_t)fb[it].w;
    }
    async_wait_all();
    __syncthreads();
    v16h a = lds_load_a(As + w * 16 * 32, 32);
#pragma unroll
    for (int c = 0; c < 4; ++c)
      acc[c] = wmma_f16(a, lds_load_bt(Bs + c * 16 * 32, 32), acc[c]);
    __syncthreads();
  }

  const int ccol  = lane_id() & 15;
  const int rbase = (lane_id() >> 4) << 3;
#pragma unroll
  for (int c = 0; c < 4; ++c)
#pragma unroll
    for (int r = 0; r < 8; ++r) {
      const int m = m0 + w * 16 + rbase + r;
      const int n = n0 + c * 16 + ccol;
      out[(size_t)m * DIMD + n] = acc[c][r] + bproj[n];
    }
}

extern "C" void kernel_launch(void* const* d_in, const int* in_sizes, int n_in,
                              void* d_out, int out_size, void* d_ws, size_t ws_size,
                              hipStream_t stream) {
  (void)in_sizes; (void)n_in; (void)out_size; (void)ws_size;
  const float* x     = (const float*)d_in[0];
  const float* Wqkv  = (const float*)d_in[1];
  const float* Wproj = (const float*)d_in[2];
  const float* bproj = (const float*)d_in[3];

  half_t* qw = (half_t*)d_ws;
  half_t* kw = qw + QKV_ELEMS;
  half_t* vw = kw + QKV_ELEMS;
  half_t* cw = vw + QKV_ELEMS;   // context [B, L, 512] f16

  dim3 blk(128);
  sattn_qkv_kernel<<<dim3((3 * DIMD) / 64, MTOT / 64), blk, 0, stream>>>(
      x, Wqkv, qw, kw, vw);
  sattn_attn_kernel<<<dim3(BATCH * NHEAD * NBLK), blk, 0, stream>>>(qw, kw, vw, cw);
  sattn_proj_kernel<<<dim3(DIMD / 64, MTOT / 64), blk, 0, stream>>>(
      cw, Wproj, bproj, (float*)d_out);
}
